// GPTPrefixSharedMLPModel_56985626084004
// MI455X (gfx1250) — matrix operand order repaired and checked by reading, not compiled, not run on hardware
//
#include <hip/hip_runtime.h>
#include <hip/hip_bf16.h>
#include <stdint.h>

// ---------------- model constants ----------------
#define BB   2
#define TT   1024
#define VV   32000
#define DD   512
#define HH   8
#define DFF  2048
#define NL   6
#define PP   64
#define HID  256
#define DK   64
#define TK   (PP + TT)        // 1088 keys (prefix + sequence)
#define MROWS (BB * TT)       // 2048

typedef __bf16 bf16;
typedef __attribute__((ext_vector_type(16))) __bf16 v16bf;
typedef __attribute__((ext_vector_type(8)))  __bf16 v8bf;
typedef __attribute__((ext_vector_type(8)))  float  v8f;

union Frag16 { v16bf v; v8bf h[2]; };

__device__ __forceinline__ v8f wmma_bf16(const Frag16& a, const Frag16& b, v8f c) {
    return __builtin_amdgcn_wmma_f32_16x16x32_bf16(false, a.v, false, b.v,
                                                   (short)0, c, false, false);
}

// ---------------- convert / transpose ----------------
// W [K x N] fp32 row-major  ->  WT [N x K] bf16 row-major
__global__ __launch_bounds__(256) void k_cvtT(const float* __restrict__ W,
                                              bf16* __restrict__ WT,
                                              int K, int N, long long total) {
    long long i = (long long)blockIdx.x * blockDim.x + threadIdx.x;
    if (i >= total) return;
    long long k = i / N;
    long long n = i - k * (long long)N;
    WT[n * (long long)K + k] = (bf16)W[i];
}

__global__ __launch_bounds__(256) void k_cvt(const float* __restrict__ in,
                                             bf16* __restrict__ out, long long n) {
    long long i = (long long)blockIdx.x * blockDim.x + threadIdx.x;
    if (i < n) out[i] = (bf16)in[i];
}

// ---------------- embedding + positional ----------------
__global__ __launch_bounds__(256) void k_embed(const int* __restrict__ ids,
                                               const float* __restrict__ emb,
                                               const float* __restrict__ pos,
                                               float* __restrict__ x,
                                               bf16* __restrict__ xb) {
    int row = blockIdx.x;              // 0 .. MROWS-1
    int t = row & (TT - 1);
    long long id = ids[row];
    for (int d = threadIdx.x; d < DD; d += blockDim.x) {
        float v = emb[id * DD + d] + pos[(long long)t * DD + d];
        x[(long long)row * DD + d]  = v;
        xb[(long long)row * DD + d] = (bf16)v;
    }
}

// =====================================================================
// Register-blocked WMMA GEMM: C = act(alpha * A @ BT^T + bias)
//   A  : bf16 [M x K] row-major (lda == K)
//   BT : bf16 [N x K] row-major (B transposed; ldb == K)
//   C / Cb addressed as z*sC + row*ldc + col (either may be null)
// Block = 128 threads = 4 waves in a 2x2 arrangement; block tile 64x128,
// wave tile 32x64 (2 A-frags x 4 B-frags -> 8 WMMA per 32-wide K step).
// Requires M % 64 == 0; N only needs N % 16 == 0 (column tiles guarded).
// ACT: 0=none, 1=tanh, 2=exact GELU
// =====================================================================
template <int ACT>
__global__ __launch_bounds__(128) void k_gemm_blk(
    const bf16* __restrict__ A, const bf16* __restrict__ BT,
    const float* __restrict__ bias, float* __restrict__ C, bf16* __restrict__ Cb,
    int M, int N, int K, int ldc, float alpha,
    long long sA, long long sB, long long sC)
{
    const int wave = threadIdx.x >> 5;
    const int lane = threadIdx.x & 31;
    const int mw = wave >> 1;            // 0..1
    const int nw = wave & 1;             // 0..1
    const int lg = lane >> 4;            // lane group
    const int li = lane & 15;
    const int m0 = blockIdx.y * 64 + mw * 32;
    const int n0 = blockIdx.x * 128 + nw * 64;
    const int z  = blockIdx.z;

    const bf16* Abase = A  + (long long)z * sA;
    const bf16* Bbase = BT + (long long)z * sB;

    // A fragment rows: m0+li and m0+16+li ; per-lane k offset lg*8 (interleaved 8-blocks)
    const bf16* ar0 = Abase + (long long)(m0 + li) * K + lg * 8;
    const bf16* ar1 = ar0 + (long long)16 * K;
    // B fragment cols: n0 + 16*j + li ; per-lane k offset lg*8 (interleaved 8-blocks)
    bool bok[4];
    const bf16* br[4];
    #pragma unroll
    for (int j = 0; j < 4; ++j) {
        int n = n0 + j * 16;
        bok[j] = (n < N);                          // wave-uniform
        br[j]  = Bbase + (long long)(n + li) * K + lg * 8;
    }

    v8f c[2][4] = {};
    for (int kk = 0; kk < K; kk += 32) {
        Frag16 a0, a1;
        a0.h[0] = *(const v8bf*)(ar0 + kk);
        a0.h[1] = *(const v8bf*)(ar0 + kk + 16);
        a1.h[0] = *(const v8bf*)(ar1 + kk);
        a1.h[1] = *(const v8bf*)(ar1 + kk + 16);
        #pragma unroll
        for (int j = 0; j < 4; ++j) {
            if (bok[j]) {                          // uniform branch: EXEC stays full
                Frag16 b;
                b.h[0] = *(const v8bf*)(br[j] + kk);
                b.h[1] = *(const v8bf*)(br[j] + kk + 16);
                c[0][j] = wmma_bf16(a0, b, c[0][j]);
                c[1][j] = wmma_bf16(a1, b, c[1][j]);
            }
        }
    }

    #pragma unroll
    for (int j = 0; j < 4; ++j) {
        if (!bok[j]) continue;
        const int col = n0 + j * 16 + li;
        const float bi = bias ? bias[col] : 0.0f;
        #pragma unroll
        for (int mi = 0; mi < 2; ++mi) {
            #pragma unroll
            for (int r = 0; r < 8; ++r) {
                int row = m0 + mi * 16 + r + lg * 8;
                float v = c[mi][j][r] * alpha + bi;
                if (ACT == 1) v = tanhf(v);
                if (ACT == 2) v = 0.5f * v * (1.0f + erff(v * 0.70710678118654752f));
                long long off = (long long)z * sC + (long long)row * ldc + col;
                if (C)  C[off]  = v;
                if (Cb) Cb[off] = (bf16)v;
            }
        }
    }
}

// ---------------- simple 1-tile-per-wave GEMM (used for att @ V, N == 64) ----
__global__ __launch_bounds__(128) void k_gemm_simple(
    const bf16* __restrict__ A, const bf16* __restrict__ BT,
    float* __restrict__ C, bf16* __restrict__ Cb,
    int M, int N, int K, int ldc, float alpha,
    long long sA, long long sB, long long sC)
{
    const int wave = threadIdx.x >> 5;
    const int lane = threadIdx.x & 31;
    const int nt = blockIdx.x * 4 + wave;
    const int mt = blockIdx.y;
    const int z  = blockIdx.z;
    if (nt * 16 >= N) return;
    const int lg = lane >> 4;
    const int li = lane & 15;

    const bf16* arow = A  + (long long)z * sA + (long long)(mt * 16 + li) * K + lg * 8;
    const bf16* brow = BT + (long long)z * sB + (long long)(nt * 16 + li) * K + lg * 8;

    v8f c = {};
    for (int kk = 0; kk < K; kk += 32) {
        Frag16 a, b;
        a.h[0] = *(const v8bf*)(arow + kk);
        a.h[1] = *(const v8bf*)(arow + kk + 16);
        b.h[0] = *(const v8bf*)(brow + kk);
        b.h[1] = *(const v8bf*)(brow + kk + 16);
        c = wmma_bf16(a, b, c);
    }

    const int col = nt * 16 + li;
    #pragma unroll
    for (int r = 0; r < 8; ++r) {
        int row = mt * 16 + r + lg * 8;
        float v = c[r] * alpha;
        long long off = (long long)z * sC + (long long)row * ldc + col;
        if (C)  C[off]  = v;
        if (Cb) Cb[off] = (bf16)v;
    }
}

// ---------------- head packing ----------------
__global__ __launch_bounds__(256) void k_pack_q(const bf16* __restrict__ src,
                                                bf16* __restrict__ Qb) {
    int i = blockIdx.x * 256 + threadIdx.x;         // 2^20 elements
    int d = i & 63, t = (i >> 6) & 1023, h = (i >> 16) & 7, b = i >> 19;
    Qb[i] = src[((long long)(b * TT + t)) * DD + h * DK + d];
}
__global__ __launch_bounds__(256) void k_pack_k(const bf16* __restrict__ src,
                                                bf16* __restrict__ Kc) {
    int i = blockIdx.x * 256 + threadIdx.x;
    int d = i & 63, t = (i >> 6) & 1023, h = (i >> 16) & 7, b = i >> 19;
    Kc[(((long long)(b * HH + h)) * TK + PP + t) * DK + d] =
        src[((long long)(b * TT + t)) * DD + h * DK + d];
}
__global__ __launch_bounds__(256) void k_pack_vt(const bf16* __restrict__ src,
                                                 bf16* __restrict__ Vt) {
    int i = blockIdx.x * 256 + threadIdx.x;
    int d = i & 63, t = (i >> 6) & 1023, h = (i >> 16) & 7, b = i >> 19;
    Vt[(((long long)(b * HH + h)) * DK + d) * TK + PP + t] =
        src[((long long)(b * TT + t)) * DD + h * DK + d];
}
__global__ __launch_bounds__(256) void k_prefix_fill(const float* __restrict__ pkv,
                                                     bf16* __restrict__ Kc,
                                                     bf16* __restrict__ Vt) {
    int i = blockIdx.x * 256 + threadIdx.x;         // 2*8*64*64 = 65536
    if (i >= BB * HH * PP * DK) return;
    int d = i & 63, p = (i >> 6) & 63, h = (i >> 12) & 7, b = i >> 15;
    Kc[(((long long)(b * HH + h)) * TK + p) * DK + d] =
        (bf16)pkv[p * (2 * HH * DK) + h * DK + d];
    Vt[(((long long)(b * HH + h)) * DK + d) * TK + p] =
        (bf16)pkv[p * (2 * HH * DK) + HH * DK + h * DK + d];
}

// ---------------- masked softmax (one row per block) ----------------
__global__ __launch_bounds__(256) void k_softmax(const float* __restrict__ logits,
                                                 bf16* __restrict__ att,
                                                 const unsigned char* __restrict__ kpm,
                                                 int b) {
    const int q = blockIdx.x, h = blockIdx.y, tid = threadIdx.x;
    const float* row = logits + ((long long)h * TT + q) * TK;
    bf16* orow = att + ((long long)h * TT + q) * TK;
    const bool qv = kpm[b * TT + q] != 0;

    float lv[5];
    float mx = -3.4e38f;
    #pragma unroll
    for (int it = 0; it < 5; ++it) {
        int j = tid + it * 256;
        float v = -3.4e38f;
        if (j < TK) {
            bool allow = qv && (j < PP || ((j - PP) <= q && kpm[b * TT + (j - PP)] != 0));
            v = allow ? row[j] : -1e9f;
        }
        lv[it] = v;
        mx = fmaxf(mx, v);
    }
    __shared__ float red[256];
    red[tid] = mx; __syncthreads();
    for (int s = 128; s > 0; s >>= 1) { if (tid < s) red[tid] = fmaxf(red[tid], red[tid + s]); __syncthreads(); }
    mx = red[0]; __syncthreads();

    float sum = 0.0f;
    #pragma unroll
    for (int it = 0; it < 5; ++it) {
        int j = tid + it * 256;
        if (j < TK) { lv[it] = expf(lv[it] - mx); sum += lv[it]; }
    }
    red[tid] = sum; __syncthreads();
    for (int s = 128; s > 0; s >>= 1) { if (tid < s) red[tid] += red[tid + s]; __syncthreads(); }
    const float inv = 1.0f / red[0];
    #pragma unroll
    for (int it = 0; it < 5; ++it) {
        int j = tid + it * 256;
        if (j < TK) orow[j] = (bf16)(lv[it] * inv);
    }
}

// ---------------- residual + LayerNorm (one row per block, D=512) ----------------
__global__ __launch_bounds__(256) void k_ln(const float* __restrict__ xin,
                                            const float* __restrict__ delta,
                                            const float* __restrict__ g,
                                            const float* __restrict__ bb,
                                            float* __restrict__ xout,
                                            bf16* __restrict__ xb) {
    const int row = blockIdx.x, tid = threadIdx.x;
    const long long base = (long long)row * DD;
    float v0 = xin[base + tid]       + delta[base + tid];
    float v1 = xin[base + tid + 256] + delta[base + tid + 256];
    __shared__ float red[256];
    red[tid] = v0 + v1; __syncthreads();
    for (int s = 128; s > 0; s >>= 1) { if (tid < s) red[tid] += red[tid + s]; __syncthreads(); }
    const float mean = red[0] * (1.0f / DD); __syncthreads();
    const float d0 = v0 - mean, d1 = v1 - mean;
    red[tid] = d0 * d0 + d1 * d1; __syncthreads();
    for (int s = 128; s > 0; s >>= 1) { if (tid < s) red[tid] += red[tid + s]; __syncthreads(); }
    const float rs = rsqrtf(red[0] * (1.0f / DD) + 1e-5f);
    float y0 = g[tid] * d0 * rs + bb[tid];
    float y1 = g[tid + 256] * d1 * rs + bb[tid + 256];
    xout[base + tid] = y0;       xout[base + tid + 256] = y1;
    xb[base + tid] = (bf16)y0;   xb[base + tid + 256]   = (bf16)y1;
}

// ---------------- host orchestration ----------------
static inline void launch_gemm_blk(int act, const bf16* A, const bf16* BT, const float* bias,
                                   float* C, bf16* Cb, int M, int N, int K, int ldc, float alpha,
                                   long long sA, long long sB, long long sC, int batch,
                                   hipStream_t s) {
    dim3 grid((N + 127) / 128, M / 64, batch), blk(128);
    if (act == 0)      k_gemm_blk<0><<<grid, blk, 0, s>>>(A, BT, bias, C, Cb, M, N, K, ldc, alpha, sA, sB, sC);
    else if (act == 1) k_gemm_blk<1><<<grid, blk, 0, s>>>(A, BT, bias, C, Cb, M, N, K, ldc, alpha, sA, sB, sC);
    else               k_gemm_blk<2><<<grid, blk, 0, s>>>(A, BT, bias, C, Cb, M, N, K, ldc, alpha, sA, sB, sC);
}

extern "C" void kernel_launch(void* const* d_in, const int* in_sizes, int n_in,
                              void* d_out, int out_size, void* d_ws, size_t ws_size,
                              hipStream_t stream) {
    // inputs (setup_inputs order)
    const int*           ids   = (const int*)d_in[0];
    const unsigned char* kpm   = (const unsigned char*)d_in[1];
    const float* emb_W  = (const float*)d_in[2];
    const float* pos_W  = (const float*)d_in[3];
    const float* prefix = (const float*)d_in[4];
    const float* sm_W1  = (const float*)d_in[5];
    const float* sm_b1  = (const float*)d_in[6];
    const float* sm_W2  = (const float*)d_in[7];
    const float* sm_b2  = (const float*)d_in[8];
    const float* pl_W   = (const float*)d_in[9];
    const float* pl_b   = (const float*)d_in[10];
    const float* q_W    = (const float*)d_in[11];
    const float* q_b    = (const float*)d_in[12];
    const float* k_W    = (const float*)d_in[13];
    const float* k_b    = (const float*)d_in[14];
    const float* v_W    = (const float*)d_in[15];
    const float* v_b    = (const float*)d_in[16];
    const float* o_W    = (const float*)d_in[17];
    const float* o_b    = (const float*)d_in[18];
    const float* ff_W1  = (const float*)d_in[19];
    const float* ff_b1  = (const float*)d_in[20];
    const float* ff_W2  = (const float*)d_in[21];
    const float* ff_b2  = (const float*)d_in[22];
    const float* ln1_g  = (const float*)d_in[23];
    const float* ln1_b  = (const float*)d_in[24];
    const float* ln2_g  = (const float*)d_in[25];
    const float* ln2_b  = (const float*)d_in[26];
    const float* cls_W  = (const float*)d_in[27];
    const float* cls_b  = (const float*)d_in[28];
    float* out = (float*)d_out;

    // ---- workspace carving (256B aligned) ----
    char* wp = (char*)d_ws;
    auto carve = [&](long long bytes) -> void* {
        void* r = (void*)wp;
        wp += (bytes + 255) & ~255LL;
        return r;
    };
    bf16* sm1T   = (bf16*)carve((long long)HID * DD * 2);
    bf16* sm2T   = (bf16*)carve((long long)HID * HID * 2);
    bf16* plT    = (bf16*)carve((long long)NL * (2 * HH * DK) * HID * 2);
    bf16* qT     = (bf16*)carve((long long)NL * DD * DD * 2);
    bf16* kT     = (bf16*)carve((long long)NL * DD * DD * 2);
    bf16* vT     = (bf16*)carve((long long)NL * DD * DD * 2);
    bf16* oT     = (bf16*)carve((long long)NL * DD * DD * 2);
    bf16* ff1T   = (bf16*)carve((long long)NL * DFF * DD * 2);
    bf16* ff2T   = (bf16*)carve((long long)NL * DD * DFF * 2);
    bf16* clsT   = (bf16*)carve((long long)VV * DD * 2);
    bf16* prefb  = (bf16*)carve((long long)PP * DD * 2);
    bf16* h1b    = (bf16*)carve((long long)PP * HID * 2);
    bf16* hb     = (bf16*)carve((long long)PP * HID * 2);
    float* pkv   = (float*)carve((long long)PP * (2 * HH * DK) * 4);
    float* x     = (float*)carve((long long)MROWS * DD * 4);
    bf16*  xb    = (bf16*)carve((long long)MROWS * DD * 2);
    bf16*  tmpb  = (bf16*)carve((long long)MROWS * DD * 2);
    bf16*  Qb    = (bf16*)carve((long long)BB * HH * TT * DK * 2);
    bf16*  Kc    = (bf16*)carve((long long)BB * HH * TK * DK * 2);
    bf16*  Vt    = (bf16*)carve((long long)BB * HH * DK * TK * 2);
    float* logit = (float*)carve((long long)HH * TT * TK * 4);
    bf16*  att   = (bf16*)carve((long long)HH * TT * TK * 2);
    bf16*  aoutb = (bf16*)carve((long long)MROWS * DD * 2);
    float* out2  = (float*)carve((long long)MROWS * DD * 4);
    bf16*  hhb   = (bf16*)carve((long long)MROWS * DFF * 2);
    (void)ws_size; (void)n_in; (void)in_sizes; (void)out_size;

    auto cvtT = [&](const float* W, bf16* WT, int K, int N) {
        long long tot = (long long)K * N;
        k_cvtT<<<dim3((unsigned)((tot + 255) / 256)), 256, 0, stream>>>(W, WT, K, N, tot);
    };

    // ---- weight conversion (bf16, transposed) ----
    cvtT(sm_W1, sm1T, DD, HID);
    cvtT(sm_W2, sm2T, HID, HID);
    for (int i = 0; i < NL; ++i) {
        cvtT(pl_W + (long long)i * HID * (2 * HH * DK), plT + (long long)i * (2 * HH * DK) * HID, HID, 2 * HH * DK);
        cvtT(q_W  + (long long)i * DD * DD,  qT  + (long long)i * DD * DD,  DD, DD);
        cvtT(k_W  + (long long)i * DD * DD,  kT  + (long long)i * DD * DD,  DD, DD);
        cvtT(v_W  + (long long)i * DD * DD,  vT  + (long long)i * DD * DD,  DD, DD);
        cvtT(o_W  + (long long)i * DD * DD,  oT  + (long long)i * DD * DD,  DD, DD);
        cvtT(ff_W1 + (long long)i * DD * DFF, ff1T + (long long)i * DFF * DD, DD, DFF);
        cvtT(ff_W2 + (long long)i * DFF * DD, ff2T + (long long)i * DD * DFF, DFF, DD);
    }
    cvtT(cls_W, clsT, DD, VV);
    {
        long long n = (long long)PP * DD;
        k_cvt<<<dim3((unsigned)((n + 255) / 256)), 256, 0, stream>>>(prefix, prefb, n);
    }

    // ---- embedding ----
    k_embed<<<MROWS, 256, 0, stream>>>(ids, emb_W, pos_W, x, xb);

    // ---- shared prefix MLP: h = tanh(tanh(prefix@W1+b1)@W2+b2) ----
    launch_gemm_blk(1, prefb, sm1T, sm_b1, nullptr, h1b, PP, HID, DD, HID, 1.0f, 0, 0, 0, 1, stream);
    launch_gemm_blk(1, h1b,   sm2T, sm_b2, nullptr, hb,  PP, HID, HID, HID, 1.0f, 0, 0, 0, 1, stream);

    const float scale = 0.125f;  // 1/sqrt(DK)
    for (int i = 0; i < NL; ++i) {
        // prefix KV: pkv = h @ pl_W[i] + pl_b[i]   [P, 2*H*DK] fp32
        launch_gemm_blk(0, hb, plT + (long long)i * (2 * HH * DK) * HID, pl_b + (long long)i * (2 * HH * DK),
                        pkv, nullptr, PP, 2 * HH * DK, HID, 2 * HH * DK, 1.0f, 0, 0, 0, 1, stream);

        // Q / K / V projections, packed into head layouts
        launch_gemm_blk(0, xb, qT + (long long)i * DD * DD, q_b + (long long)i * DD,
                        nullptr, tmpb, MROWS, DD, DD, DD, 1.0f, 0, 0, 0, 1, stream);
        k_pack_q<<<4096, 256, 0, stream>>>(tmpb, Qb);
        launch_gemm_blk(0, xb, kT + (long long)i * DD * DD, k_b + (long long)i * DD,
                        nullptr, tmpb, MROWS, DD, DD, DD, 1.0f, 0, 0, 0, 1, stream);
        k_pack_k<<<4096, 256, 0, stream>>>(tmpb, Kc);
        launch_gemm_blk(0, xb, vT + (long long)i * DD * DD, v_b + (long long)i * DD,
                        nullptr, tmpb, MROWS, DD, DD, DD, 1.0f, 0, 0, 0, 1, stream);
        k_pack_vt<<<4096, 256, 0, stream>>>(tmpb, Vt);
        k_prefix_fill<<<256, 256, 0, stream>>>(pkv, Kc, Vt);

        // attention per batch element, batched over heads via blockIdx.z
        for (int b = 0; b < BB; ++b) {
            // logits[h] = scale * Q[b,h] @ K[b,h]^T   (M=T=1024, N=TK=1088, K=DK=64)
            launch_gemm_blk(0, Qb + (long long)b * HH * TT * DK,
                               Kc + (long long)b * HH * TK * DK,
                               nullptr, logit, nullptr,
                               TT, TK, DK, TK, scale,
                               (long long)TT * DK, (long long)TK * DK, (long long)TT * TK,
                               HH, stream);
            k_softmax<<<dim3(TT, HH), 256, 0, stream>>>(logit, att, kpm, b);
            // attout[b,t, h*DK + d] = att[h] @ V[b,h]   (M=T, N=DK=64, K=TK)
            {
                dim3 grid(1, TT / 16, HH), blk(128);
                k_gemm_simple<<<grid, blk, 0, stream>>>(
                    att, Vt + (long long)b * HH * DK * TK,
                    nullptr, aoutb + (long long)b * TT * DD,
                    TT, DK, TK, DD, 1.0f,
                    (long long)TT * TK, (long long)DK * TK, (long long)DK);
            }
        }

        // output projection + residual + LN1
        launch_gemm_blk(0, aoutb, oT + (long long)i * DD * DD, o_b + (long long)i * DD,
                        out2, nullptr, MROWS, DD, DD, DD, 1.0f, 0, 0, 0, 1, stream);
        k_ln<<<MROWS, 256, 0, stream>>>(x, out2, ln1_g + (long long)i * DD, ln1_b + (long long)i * DD, x, xb);

        // FFN: gelu(x@W1+b1)@W2+b2, residual + LN2
        launch_gemm_blk(2, xb, ff1T + (long long)i * DFF * DD, ff_b1 + (long long)i * DFF,
                        nullptr, hhb, MROWS, DFF, DD, DFF, 1.0f, 0, 0, 0, 1, stream);
        launch_gemm_blk(0, hhb, ff2T + (long long)i * DD * DFF, ff_b2 + (long long)i * DD,
                        out2, nullptr, MROWS, DD, DFF, DD, 1.0f, 0, 0, 0, 1, stream);
        k_ln<<<MROWS, 256, 0, stream>>>(x, out2, ln2_g + (long long)i * DD, ln2_b + (long long)i * DD, x, xb);
    }

    // ---- classifier: out = x @ cls_W + cls_b  (fp32 out, M=2048, N=32000, K=512) ----
    launch_gemm_blk(0, xb, clsT, cls_b, out, nullptr, MROWS, VV, DD, VV, 1.0f, 0, 0, 0, 1, stream);
}
